// CausalSelfAttention_28647431864393
// MI455X (gfx1250) — matrix operand order repaired and checked by reading, not compiled, not run on hardware
//
#include <hip/hip_runtime.h>

typedef __attribute__((ext_vector_type(16))) _Float16 v16h;
typedef __attribute__((ext_vector_type(8)))  _Float16 v8h;
typedef __attribute__((ext_vector_type(8)))  float    v8f;

#define TSEQ 4096
#define EMB  1024
#define NH   16
#define DH   64

static __device__ __forceinline__ v8f vzero8f() {
  v8f z = {0.f, 0.f, 0.f, 0.f, 0.f, 0.f, 0.f, 0.f};
  return z;
}

// A operand (16x32 f16) for v_wmma_f32_16x16x32_f16 from row-major f16 src.
// Lane = row (lane&15); lane-half selects K sub-range: elems i -> K = 2*h8 + i
// (h8 = (lane>>4)*8), i.e. K = (lane>>4)*16 + i.
static __device__ __forceinline__ v16h load_A_rm(const _Float16* __restrict__ p,
                                                 int ld, int m0, int k0) {
  int lane = threadIdx.x & 31;
  int r  = lane & 15;
  int h8 = (lane >> 4) << 3;
  const _Float16* rp = p + (size_t)(m0 + r) * ld + k0;
  union { v16h v; v8h g[2]; } u;
  u.g[0] = *(const v8h*)(rp + 2 * h8);
  u.g[1] = *(const v8h*)(rp + 2 * h8 + 8);
  return u.v;
}

// B operand (32x16 f16): source stored N-major, p[n*ld + k] (K contiguous).
// Lane = column (lane&15); elems i -> K = k0 + (lane>>4)*16 + i.
static __device__ __forceinline__ v16h load_B_nm(const _Float16* __restrict__ p,
                                                 int ld, int k0, int n0) {
  int lane = threadIdx.x & 31;
  int c   = lane & 15;
  int h16 = (lane >> 4) << 4;
  const _Float16* cp = p + (size_t)(n0 + c) * ld + k0 + h16;
  union { v16h v; v8h g[2]; } u;
  u.g[0] = *(const v8h*)(cp);
  u.g[1] = *(const v8h*)(cp + 8);
  return u.v;
}

// B operand from an LDS-staged row-major 16x32 f16 tile (row n at byte n*64).
static __device__ __forceinline__ v16h load_B_lds(const _Float16* p) {
  int lane = threadIdx.x & 31;
  int c   = lane & 15;
  int h16 = (lane >> 4) << 4;
  const _Float16* cp = p + c * 32 + h16;
  union { v16h v; v8h g[2]; } u;
  u.g[0] = *(const v8h*)(cp);
  u.g[1] = *(const v8h*)(cp + 8);
  return u.v;
}

#define WMMA_F32_F16(a, b, c) \
  __builtin_amdgcn_wmma_f32_16x16x32_f16(false, (a), false, (b), (short)0, (c), false, false)

// Stage one 16x32 f16 B tile (1 KB) global->LDS with two async b128 ops.
// Lane i copies 16 B: instr0 covers rows 0..7, instr1 rows 8..15.
// src points at the tile origin (row n0, col k0); rows stride EMB halfs.
static __device__ __forceinline__ void stage_B_async(unsigned ldsTileBase,
                                                     const _Float16* src,
                                                     unsigned va0, unsigned va1,
                                                     unsigned ldl) {
  unsigned long long sa = (unsigned long long)(uintptr_t)src;
  asm volatile("global_load_async_to_lds_b128 %0, %1, %2 offset:0"
               :: "v"(ldsTileBase + ldl), "v"(va0), "s"(sa) : "memory");
  asm volatile("global_load_async_to_lds_b128 %0, %1, %2 offset:0"
               :: "v"(ldsTileBase + 512 + ldl), "v"(va1), "s"(sa) : "memory");
}

// ---------------------------------------------------------------- prep kernels
__global__ void cvt_f32_f16_kernel(const float* __restrict__ src,
                                   _Float16* __restrict__ dst, int n) {
  int i = blockIdx.x * blockDim.x + threadIdx.x;
  if (i < n) dst[i] = (_Float16)src[i];
}

// W[k*N + n] (f32, K x N) -> Wt[n*K + k] (f16, N-major)
__global__ void transpose_cvt_kernel(const float* __restrict__ W,
                                     _Float16* __restrict__ Wt, int K, int N) {
  int i = blockIdx.x * blockDim.x + threadIdx.x;
  if (i < K * N) {
    int k = i / N, n = i - k * N;
    Wt[(size_t)n * K + k] = (_Float16)W[i];
  }
}

// ---------------------------------------------------------------- QKV GEMM
// Per wave: C[32 x 64] (2 M-tiles x 4 N-tiles), B double-buffered through LDS via
// global_load_async_to_lds_b128 (ASYNCcnt pipeline). Scatter into Q[h][t][64] (x 1/8),
// K[h][t][64], Vt[h][64][t].
__global__ __launch_bounds__(128) void qkv_gemm_kernel(
    const _Float16* __restrict__ Xh, const _Float16* __restrict__ Wt,
    const float* __restrict__ bias,
    _Float16* __restrict__ Qh, _Float16* __restrict__ Kh, _Float16* __restrict__ VhT) {
  __shared__ __align__(16) _Float16 ldsB[4][2][4][16 * 32];  // 32 KB
  int wave = threadIdx.x >> 5;
  int lane = threadIdx.x & 31;
  int m0 = blockIdx.x << 5;
  int nc = (blockIdx.y << 8) + (wave << 6);

  const _Float16* bj[4];
#pragma unroll
  for (int j = 0; j < 4; ++j) bj[j] = Wt + (size_t)(nc + (j << 4)) * EMB;

  // LDS byte offset of this wave's staging area (flat addr low 32 bits = LDS offset)
  unsigned ldsWave = (unsigned)(uintptr_t)(&ldsB[wave][0][0][0]);
  unsigned inrow = (lane & 3) << 4;                 // byte within 64-B tile row
  unsigned row0  = (unsigned)(lane >> 2);           // tile row 0..7
  unsigned va0 = row0 * (EMB * 2) + inrow;          // global byte offset, rows 0..7
  unsigned va1 = (row0 + 8) * (EMB * 2) + inrow;    // rows 8..15
  unsigned ldl = (unsigned)lane << 4;               // lane's 16-B slot

  // prologue: stage kc=0 into buffer 0
#pragma unroll
  for (int j = 0; j < 4; ++j)
    stage_B_async(ldsWave + (j << 10), bj[j], va0, va1, ldl);

  v8f acc[2][4];
#pragma unroll
  for (int mi = 0; mi < 2; ++mi)
#pragma unroll
    for (int j = 0; j < 4; ++j) acc[mi][j] = vzero8f();

  int buf = 0;
  for (int kc = 0; kc < EMB; kc += 32) {
    if (kc + 32 < EMB) {
      asm volatile("s_wait_dscnt 0x0" ::: "memory");  // prior reads of buf^1 done
      unsigned bb = ldsWave + ((unsigned)(buf ^ 1) << 12);
#pragma unroll
      for (int j = 0; j < 4; ++j)
        stage_B_async(bb + (j << 10), bj[j] + kc + 32, va0, va1, ldl);
      asm volatile("s_wait_asynccnt 0x8" ::: "memory");  // current batch landed
    } else {
      asm volatile("s_wait_asynccnt 0x0" ::: "memory");
    }
    v16h a0 = load_A_rm(Xh, EMB, m0, kc);
    v16h a1 = load_A_rm(Xh, EMB, m0 + 16, kc);
#pragma unroll
    for (int j = 0; j < 4; ++j) {
      v16h b = load_B_lds(&ldsB[wave][buf][j][0]);
      acc[0][j] = WMMA_F32_F16(a0, b, acc[0][j]);
      acc[1][j] = WMMA_F32_F16(a1, b, acc[1][j]);
    }
    buf ^= 1;
  }

  int col = lane & 15, hf = lane >> 4;
#pragma unroll
  for (int mi = 0; mi < 2; ++mi)
#pragma unroll
    for (int j = 0; j < 4; ++j) {
      int n = nc + (j << 4) + col;
      float bv = bias[n];
      int which = n >> 10;
      int e = n & 1023;
      int h = e >> 6, d = e & 63;
#pragma unroll
      for (int v = 0; v < 8; ++v) {
        int t = m0 + (mi << 4) + v + (hf << 3);
        float val = acc[mi][j][v] + bv;
        if (which == 0)      Qh[((size_t)h * TSEQ + t) * DH + d] = (_Float16)(val * 0.125f);
        else if (which == 1) Kh[((size_t)h * TSEQ + t) * DH + d] = (_Float16)val;
        else                 VhT[((size_t)h * DH + d) * TSEQ + t] = (_Float16)val;
      }
    }
}

// ---------------------------------------------------------------- flash attention
// One wave per 16-query tile per head; 32-key chunks; online softmax in f32.
__global__ __launch_bounds__(128) void attn_kernel(
    const _Float16* __restrict__ Qh, const _Float16* __restrict__ Kh,
    const _Float16* __restrict__ VhT, _Float16* __restrict__ AttnH) {
  __shared__ __align__(16) _Float16 ldsP[4][16 * 32];
  int wave = threadIdx.x >> 5;
  int lane = threadIdx.x & 31;
  int col = lane & 15, hf = lane >> 4;
  int h = blockIdx.y;
  int q0 = ((blockIdx.x << 2) + wave) << 4;
  const _Float16* Q = Qh + (size_t)h * TSEQ * DH;
  const _Float16* K = Kh + (size_t)h * TSEQ * DH;
  const _Float16* V = VhT + (size_t)h * DH * TSEQ;
  _Float16* P = ldsP[wave];

  v16h aq0 = load_A_rm(Q, DH, q0, 0);
  v16h aq1 = load_A_rm(Q, DH, q0, 32);

  float m[8], l[8];
  v8f o[4];
#pragma unroll
  for (int v = 0; v < 8; ++v) { m[v] = -__builtin_inff(); l[v] = 0.f; }
#pragma unroll
  for (int j = 0; j < 4; ++j) o[j] = vzero8f();

  int qmax = q0 + 15;
  for (int kb = 0; kb <= qmax; kb += 32) {
    // S = Q K^T (Q pre-scaled by 1/sqrt(D)), 2 key-subtiles x 2 d-chunks
    v8f s[2];
#pragma unroll
    for (int nt = 0; nt < 2; ++nt) {
      v16h b0 = load_B_nm(K, DH, 0,  kb + (nt << 4));
      v16h b1 = load_B_nm(K, DH, 32, kb + (nt << 4));
      v8f z = vzero8f();
      z = WMMA_F32_F16(aq0, b0, z);
      s[nt] = WMMA_F32_F16(aq1, b1, z);
    }
    // causal mask + per-row chunk max (row = vgpr idx + 8*half, col across 16 lanes)
    float cm[8];
#pragma unroll
    for (int v = 0; v < 8; ++v) cm[v] = -1e30f;
#pragma unroll
    for (int nt = 0; nt < 2; ++nt) {
      int key = kb + (nt << 4) + col;
#pragma unroll
      for (int v = 0; v < 8; ++v) {
        int q = q0 + v + (hf << 3);
        if (key > q) s[nt][v] = -1e30f;
        else cm[v] = fmaxf(cm[v], s[nt][v]);
      }
    }
#pragma unroll
    for (int mk = 1; mk < 16; mk <<= 1)
#pragma unroll
      for (int v = 0; v < 8; ++v) cm[v] = fmaxf(cm[v], __shfl_xor(cm[v], mk, 32));

    float alpha[8], rs[8];
#pragma unroll
    for (int v = 0; v < 8; ++v) {
      float mn = fmaxf(m[v], cm[v]);
      alpha[v] = __expf(m[v] - mn);
      m[v] = mn;
      rs[v] = 0.f;
    }
#pragma unroll
    for (int nt = 0; nt < 2; ++nt)
#pragma unroll
      for (int v = 0; v < 8; ++v) {
        float pe = __expf(s[nt][v] - m[v]);
        s[nt][v] = pe;
        rs[v] += pe;
      }
#pragma unroll
    for (int mk = 1; mk < 16; mk <<= 1)
#pragma unroll
      for (int v = 0; v < 8; ++v) rs[v] += __shfl_xor(rs[v], mk, 32);
#pragma unroll
    for (int v = 0; v < 8; ++v) l[v] = l[v] * alpha[v] + rs[v];
#pragma unroll
    for (int j = 0; j < 4; ++j)
#pragma unroll
      for (int v = 0; v < 8; ++v) o[j][v] *= alpha[v];

    // P: C-layout -> A-layout via private LDS slice (in-order DS, wave-local)
#pragma unroll
    for (int nt = 0; nt < 2; ++nt)
#pragma unroll
      for (int v = 0; v < 8; ++v)
        P[(v + (hf << 3)) * 32 + (nt << 4) + col] = (_Float16)s[nt][v];
    asm volatile("s_wait_dscnt 0x0" ::: "memory");
    union { v16h vv; v8h g[2]; } ua;
    ua.g[0] = *(const v8h*)&P[col * 32 + (hf << 4)];
    ua.g[1] = *(const v8h*)&P[col * 32 + (hf << 4) + 8];
    v16h pa = ua.vv;

    // O += P V  (V stored d-major so it's a clean B operand)
#pragma unroll
    for (int j = 0; j < 4; ++j) {
      v16h bv = load_B_nm(V, TSEQ, kb, j << 4);
      o[j] = WMMA_F32_F16(pa, bv, o[j]);
    }
  }
#pragma unroll
  for (int v = 0; v < 8; ++v) l[v] = 1.f / l[v];
#pragma unroll
  for (int j = 0; j < 4; ++j)
#pragma unroll
    for (int v = 0; v < 8; ++v) {
      int t = q0 + v + (hf << 3);
      AttnH[(size_t)t * EMB + h * DH + (j << 4) + col] = (_Float16)(o[j][v] * l[v]);
    }
}

// ---------------------------------------------------------------- output projection
__global__ __launch_bounds__(128) void out_gemm_kernel(
    const _Float16* __restrict__ Ah, const _Float16* __restrict__ Wt,
    const float* __restrict__ bias, float* __restrict__ out) {
  __shared__ __align__(16) _Float16 ldsB[4][2][4][16 * 32];  // 32 KB
  int wave = threadIdx.x >> 5;
  int lane = threadIdx.x & 31;
  int m0 = blockIdx.x << 5;
  int nc = (blockIdx.y << 8) + (wave << 6);

  const _Float16* bj[4];
#pragma unroll
  for (int j = 0; j < 4; ++j) bj[j] = Wt + (size_t)(nc + (j << 4)) * EMB;

  unsigned ldsWave = (unsigned)(uintptr_t)(&ldsB[wave][0][0][0]);
  unsigned inrow = (lane & 3) << 4;
  unsigned row0  = (unsigned)(lane >> 2);
  unsigned va0 = row0 * (EMB * 2) + inrow;
  unsigned va1 = (row0 + 8) * (EMB * 2) + inrow;
  unsigned ldl = (unsigned)lane << 4;

#pragma unroll
  for (int j = 0; j < 4; ++j)
    stage_B_async(ldsWave + (j << 10), bj[j], va0, va1, ldl);

  v8f acc[2][4];
#pragma unroll
  for (int mi = 0; mi < 2; ++mi)
#pragma unroll
    for (int j = 0; j < 4; ++j) acc[mi][j] = vzero8f();

  int buf = 0;
  for (int kc = 0; kc < EMB; kc += 32) {
    if (kc + 32 < EMB) {
      asm volatile("s_wait_dscnt 0x0" ::: "memory");
      unsigned bb = ldsWave + ((unsigned)(buf ^ 1) << 12);
#pragma unroll
      for (int j = 0; j < 4; ++j)
        stage_B_async(bb + (j << 10), bj[j] + kc + 32, va0, va1, ldl);
      asm volatile("s_wait_asynccnt 0x8" ::: "memory");
    } else {
      asm volatile("s_wait_asynccnt 0x0" ::: "memory");
    }
    v16h a0 = load_A_rm(Ah, EMB, m0, kc);
    v16h a1 = load_A_rm(Ah, EMB, m0 + 16, kc);
#pragma unroll
    for (int j = 0; j < 4; ++j) {
      v16h b = load_B_lds(&ldsB[wave][buf][j][0]);
      acc[0][j] = WMMA_F32_F16(a0, b, acc[0][j]);
      acc[1][j] = WMMA_F32_F16(a1, b, acc[1][j]);
    }
    buf ^= 1;
  }

  int col = lane & 15, hf = lane >> 4;
#pragma unroll
  for (int mi = 0; mi < 2; ++mi)
#pragma unroll
    for (int j = 0; j < 4; ++j) {
      int n = nc + (j << 4) + col;
      float bv = bias[n];
#pragma unroll
      for (int v = 0; v < 8; ++v) {
        int t = m0 + (mi << 4) + v + (hf << 3);
        out[(size_t)t * EMB + n] = acc[mi][j][v] + bv;
      }
    }
}

extern "C" void kernel_launch(void* const* d_in, const int* in_sizes, int n_in,
                              void* d_out, int out_size, void* d_ws, size_t ws_size,
                              hipStream_t stream) {
  (void)in_sizes; (void)n_in; (void)out_size; (void)ws_size;
  const float* x     = (const float*)d_in[0];
  const float* w_qkv = (const float*)d_in[1];
  const float* b_qkv = (const float*)d_in[2];
  const float* w_out = (const float*)d_in[3];
  const float* b_out = (const float*)d_in[4];
  float* out = (float*)d_out;

  char* ws = (char*)d_ws;
  const size_t TE = (size_t)TSEQ * EMB;           // 4M elems
  _Float16* Xh    = (_Float16*)(ws);                                       // 8 MB
  _Float16* WqkvT = (_Float16*)(ws + TE * 2);                              // 6 MB
  _Float16* WoutT = (_Float16*)(ws + TE * 2 + (size_t)3 * EMB * EMB * 2);  // 2 MB
  _Float16* Qh    = (_Float16*)(ws + TE * 2 + (size_t)4 * EMB * EMB * 2);  // 8 MB
  _Float16* Kh    = Qh + (size_t)NH * TSEQ * DH;                           // 8 MB
  _Float16* VhT   = Kh + (size_t)NH * TSEQ * DH;                           // 8 MB
  _Float16* AttnH = Xh;  // x-f16 buffer dead after QKV GEMM; reuse

  cvt_f32_f16_kernel<<<(TSEQ * EMB) / 256, 256, 0, stream>>>(x, Xh, TSEQ * EMB);
  transpose_cvt_kernel<<<(EMB * 3 * EMB) / 256, 256, 0, stream>>>(w_qkv, WqkvT, EMB, 3 * EMB);
  transpose_cvt_kernel<<<(EMB * EMB) / 256, 256, 0, stream>>>(w_out, WoutT, EMB, EMB);

  qkv_gemm_kernel<<<dim3(TSEQ / 32, (3 * EMB) / 256), 128, 0, stream>>>(
      Xh, WqkvT, b_qkv, Qh, Kh, VhT);

  attn_kernel<<<dim3(TSEQ / 64, NH), 128, 0, stream>>>(Qh, Kh, VhT, AttnH);

  out_gemm_kernel<<<dim3(TSEQ / 32, EMB / 256), 128, 0, stream>>>(
      AttnH, WoutT, b_out, out);
}
